// GraphSAGE_11012296147627
// MI455X (gfx1250) — matrix operand order repaired and checked by reading, not compiled, not run on hardware
//
#include <hip/hip_runtime.h>
#include <hip/hip_bf16.h>

// ---------------------------------------------------------------------------
// GraphSAGE (2-layer, mean aggregation) for MI455X / gfx1250, wave32 + WMMA.
//
// Geometry: N0=1048576, N1=65536, N2=4096, DEG0=16, DEG1=10,
//           D_IN=128, D_H=256, D_OUT=47.
//
// Pipeline (all on `stream`, dependencies via stream order):
//   1) packB: Wl/Wr (f32) -> WMMA-lane-ordered bf16 hi/lo B tiles
//   2) agg16: mean over 16 neighbors of x            -> mean0 [N1,128] f32
//   3) packA: concat(mean0, x[:N1]) -> bf16 hi/lo A tiles (K=256)
//   4) gemm : A0 x B0 + bl0, GELU                    -> h0 [N1,256] f32
//   5) agg10: mean over 10 neighbors of h0           -> mean1 [N2,256]
//   6) packA: concat(mean1, h0[:N2]) -> A1 (K=512)
//   7) gemm : A1 x B1 + bl1, GELU                    -> h1 [N2,256]
//   8) packA: h1 -> A2 (K=256)
//   9) gemm : A2 x B2(Wo padded to N=48) + bo        -> out [N2,47]
//
// Precision: bf16x3 split (hi + lo residual; AhBh + AhBl + AlBh, f32 accum)
// gives ~2^-15 relative error vs the f32 reference while running on the
// fast v_wmma_f32_16x16x32_bf16 pipe.
// ---------------------------------------------------------------------------

typedef __attribute__((ext_vector_type(16))) __bf16 v16bf;
typedef __attribute__((ext_vector_type(8)))  float  v8f;
typedef __attribute__((ext_vector_type(4)))  float  v4f;

// ---------------- neighbor-mean kernels (1 wave per dst row) ----------------

__global__ void agg_mean16_kernel(const float* __restrict__ x,
                                  const int* __restrict__ idx,
                                  float* __restrict__ out, int ndst) {
  const int lane = threadIdx.x & 31;
  const int j = (blockIdx.x * blockDim.x + threadIdx.x) >> 5;
  if (j >= ndst) return;
  int myidx = idx[j * 16 + (lane & 15)];
  v4f acc = {0.f, 0.f, 0.f, 0.f};
  #pragma unroll
  for (int k = 0; k < 16; ++k) {
    int src = __shfl(myidx, k, 32);
    const v4f* row = reinterpret_cast<const v4f*>(x + (size_t)src * 128);
    acc += row[lane];                       // 32 lanes x 16B = one 512B row
  }
  acc *= (1.0f / 16.0f);
  reinterpret_cast<v4f*>(out + (size_t)j * 128)[lane] = acc;
}

__global__ void agg_mean10_kernel(const float* __restrict__ h,
                                  const int* __restrict__ idx,
                                  float* __restrict__ out, int ndst) {
  const int lane = threadIdx.x & 31;
  const int j = (blockIdx.x * blockDim.x + threadIdx.x) >> 5;
  if (j >= ndst) return;
  int myidx = idx[j * 10 + (lane < 10 ? lane : 9)];
  v4f a0 = {0.f, 0.f, 0.f, 0.f}, a1 = {0.f, 0.f, 0.f, 0.f};
  #pragma unroll
  for (int k = 0; k < 10; ++k) {
    int src = __shfl(myidx, k, 32);
    const v4f* row = reinterpret_cast<const v4f*>(h + (size_t)src * 256);
    a0 += row[lane];
    a1 += row[32 + lane];
  }
  a0 *= 0.1f; a1 *= 0.1f;
  v4f* o = reinterpret_cast<v4f*>(out + (size_t)j * 256);
  o[lane] = a0; o[32 + lane] = a1;
}

// ---------------- operand packing into WMMA lane order ----------------------
// 16-bit A (16x32, MxK) per ISA: lane L holds row M=L&15;
//   VGPR i (i<4):  K = (L>>4)*8 + {2i,2i+1};  VGPR i (i>=4): K = 16 + same.
// => element j of v16bf:  k = ((j<8)?0:16) + (L>>4)*8 + (j&7)
// 16-bit B (32x16, KxN): lane L holds col N=L&15;
//   same K order as A => element j: k = ((j<8)?0:16) + (L>>4)*8 + (j&7)
// Tile storage: [tile][lane][32 bf16]  (hi[0:16], lo[16:32]) -> 64B/lane.

__global__ void packA_kernel(const float* __restrict__ S0, int K0,
                             const float* __restrict__ S1, int K1,
                             __bf16* __restrict__ P, int KT, int ntiles) {
  int t = blockIdx.x * blockDim.x + threadIdx.x;   // one thread per (tile,lane)
  int lane = t & 31;
  int tile = t >> 5;
  if (tile >= ntiles) return;
  int kt = tile % KT;
  int mt = tile / KT;
  int m = mt * 16 + (lane & 15);
  int khalf = (lane >> 4) * 8;
  alignas(16) __bf16 o[32];
  #pragma unroll
  for (int j = 0; j < 16; ++j) {
    int k = kt * 32 + ((j < 8) ? 0 : 16) + khalf + (j & 7);
    float f = (k < K0) ? S0[(size_t)m * K0 + k]
                       : S1[(size_t)m * K1 + (k - K0)];
    __bf16 hi = (__bf16)f;
    __bf16 lo = (__bf16)(f - (float)hi);
    o[j] = hi; o[16 + j] = lo;
  }
  uint4* dst = reinterpret_cast<uint4*>(P + (size_t)t * 32);
  const uint4* src = reinterpret_cast<const uint4*>(o);
  dst[0] = src[0]; dst[1] = src[1]; dst[2] = src[2]; dst[3] = src[3];
}

__global__ void packB_kernel(const float* __restrict__ Wl, int K0,
                             const float* __restrict__ Wr, int K1,
                             int Nreal,
                             __bf16* __restrict__ P, int NTtot, int ntiles) {
  int t = blockIdx.x * blockDim.x + threadIdx.x;
  int lane = t & 31;
  int tile = t >> 5;
  if (tile >= ntiles) return;
  int nt = tile % NTtot;
  int kt = tile / NTtot;
  int n = nt * 16 + (lane & 15);
  int kb = (lane >> 4) * 8;
  alignas(16) __bf16 o[32];
  #pragma unroll
  for (int j = 0; j < 16; ++j) {
    int k = kt * 32 + ((j < 8) ? 0 : 16) + kb + (j & 7);
    float f = 0.f;
    if (n < Nreal)
      f = (k < K0) ? Wl[(size_t)n * K0 + k] : Wr[(size_t)n * K1 + (k - K0)];
    __bf16 hi = (__bf16)f;
    __bf16 lo = (__bf16)(f - (float)hi);
    o[j] = hi; o[16 + j] = lo;
  }
  uint4* dst = reinterpret_cast<uint4*>(P + (size_t)t * 32);
  const uint4* src = reinterpret_cast<const uint4*>(o);
  dst[0] = src[0]; dst[1] = src[1]; dst[2] = src[2]; dst[3] = src[3];
}

// ---------------- packed bf16x3 WMMA GEMM + bias (+ exact GELU) -------------
// 1 wave computes a 16-row M-strip x (NT*16)-col N-slab. block = 128 (4 waves).
// Grids are exact multiples -> EXEC is all-1s through every WMMA.

template <int NT, bool GELU>
__global__ void gemm_kernel(const __bf16* __restrict__ Ap,
                            const __bf16* __restrict__ Bp,
                            const float* __restrict__ bias,
                            float* __restrict__ C,
                            int Nreal, int KT, int NTtot) {
  const int lane = threadIdx.x & 31;
  const int wave = threadIdx.x >> 5;
  const int mt = blockIdx.x * 4 + wave;
  const int nt0 = blockIdx.y * NT;

  v8f acc[NT];
  #pragma unroll
  for (int i = 0; i < NT; ++i)
    acc[i] = (v8f){0.f, 0.f, 0.f, 0.f, 0.f, 0.f, 0.f, 0.f};

  // tile = 32 lanes * 32 bf16 = 64 v16bf; lane's slice is 2 v16bf (hi, lo)
  const v16bf* abase =
      reinterpret_cast<const v16bf*>(Ap) + (size_t)mt * KT * 64 + lane * 2;
  const v16bf* bbase0 = reinterpret_cast<const v16bf*>(Bp) + lane * 2;

  for (int kt = 0; kt < KT; ++kt) {
    v16bf a_hi = abase[(size_t)kt * 64];
    v16bf a_lo = abase[(size_t)kt * 64 + 1];
    const v16bf* bb = bbase0 + (size_t)(kt * NTtot + nt0) * 64;
    #pragma unroll
    for (int nt = 0; nt < NT; ++nt) {
      v16bf b_hi = bb[(size_t)nt * 64];
      v16bf b_lo = bb[(size_t)nt * 64 + 1];
      acc[nt] = __builtin_amdgcn_wmma_f32_16x16x32_bf16(
          false, a_hi, false, b_hi, (short)0, acc[nt], false, false);
      acc[nt] = __builtin_amdgcn_wmma_f32_16x16x32_bf16(
          false, a_hi, false, b_lo, (short)0, acc[nt], false, false);
      acc[nt] = __builtin_amdgcn_wmma_f32_16x16x32_bf16(
          false, a_lo, false, b_hi, (short)0, acc[nt], false, false);
    }
  }

  // C/D layout: VGPR v -> row v (lanes 0-15) / row 8+v (lanes 16-31), col=lane&15
  const int rbase = mt * 16 + ((lane >> 4) << 3);
  const int cl = lane & 15;
  #pragma unroll
  for (int nt = 0; nt < NT; ++nt) {
    int c = (nt0 + nt) * 16 + cl;
    if (c >= Nreal) continue;
    float bv = bias[c];
    #pragma unroll
    for (int v = 0; v < 8; ++v) {
      float val = acc[nt][v] + bv;
      if (GELU) val = 0.5f * val * (1.0f + erff(val * 0.70710678118654752f));
      C[(size_t)(rbase + v) * Nreal + c] = val;
    }
  }
}

// ---------------------------------------------------------------------------

extern "C" void kernel_launch(void* const* d_in, const int* in_sizes, int n_in,
                              void* d_out, int out_size, void* d_ws, size_t ws_size,
                              hipStream_t stream) {
  (void)in_sizes; (void)n_in; (void)out_size; (void)ws_size;

  const float* x    = (const float*)d_in[0];    // [1048576,128]
  const int*   idx0 = (const int*)  d_in[1];    // [1048576]
  const int*   idx1 = (const int*)  d_in[3];    // [40960]
  const float* Wl0  = (const float*)d_in[5];    // [256,128]
  const float* bl0  = (const float*)d_in[6];    // [256]
  const float* Wr0  = (const float*)d_in[7];    // [256,128]
  const float* Wl1  = (const float*)d_in[8];    // [256,256]
  const float* bl1  = (const float*)d_in[9];    // [256]
  const float* Wr1  = (const float*)d_in[10];   // [256,256]
  const float* Wo   = (const float*)d_in[11];   // [47,256]
  const float* bo   = (const float*)d_in[12];   // [47]
  float* out = (float*)d_out;                   // [4096,47]

  char* ws = (char*)d_ws;
  const size_t MB = 1024u * 1024u;
  // Region [0,32MB): mean0 first; reused after pack_A0 for the tiny layer-1
  // buffers (liveness does not overlap).
  float*  mean0 = (float*)(ws);                 // 32 MB  [65536,128]
  float*  mean1 = (float*)(ws);                 //  4 MB  [4096,256]
  __bf16* A1p   = (__bf16*)(ws + 4  * MB);      //  8 MB
  float*  h1    = (float*)(ws + 12 * MB);       //  4 MB  [4096,256]
  __bf16* A2p   = (__bf16*)(ws + 16 * MB);      //  4 MB
  float*  h0    = (float*)(ws + 32 * MB);       // 64 MB  [65536,256]
  __bf16* A0p   = (__bf16*)(ws + 96 * MB);      // 64 MB
  __bf16* B0p   = (__bf16*)(ws + 160 * MB);     // 256 KB
  __bf16* B1p   = (__bf16*)(ws + 161 * MB);     // 512 KB
  __bf16* B2p   = (__bf16*)(ws + 162 * MB);     //  48 KB

  // ---- weight packing (tiny) ----
  // B0: K=128+128, N=256 -> KT=8, NT=16, tiles=128
  packB_kernel<<<16, 256, 0, stream>>>(Wl0, 128, Wr0, 128, 256, B0p, 16, 128);
  // B1: K=256+256, N=256 -> KT=16, NT=16, tiles=256
  packB_kernel<<<32, 256, 0, stream>>>(Wl1, 256, Wr1, 256, 256, B1p, 16, 256);
  // B2: K=256, Nreal=47 padded to 48 -> KT=8, NT=3, tiles=24
  packB_kernel<<<3, 256, 0, stream>>>(Wo, 256, Wo, 0, 47, B2p, 3, 24);

  // ---- layer 0 ----
  agg_mean16_kernel<<<8192, 256, 0, stream>>>(x, idx0, mean0, 65536);
  // A0 = [mean0 | x[:N1]] : M=65536, K=256 -> KT=8, tiles=4096*8=32768
  packA_kernel<<<4096, 256, 0, stream>>>(mean0, 128, x, 128, A0p, 8, 32768);
  // h0 = GELU(A0 @ B0 + bl0): 4096 M-strips / 4 waves, N split in 2 slabs
  gemm_kernel<8, true><<<dim3(1024, 2), 128, 0, stream>>>(
      A0p, B0p, bl0, h0, 256, 8, 16);

  // ---- layer 1 ----
  agg_mean10_kernel<<<512, 256, 0, stream>>>(h0, idx1, mean1, 4096);
  // A1 = [mean1 | h0[:N2]] : M=4096, K=512 -> KT=16, tiles=256*16=4096
  packA_kernel<<<512, 256, 0, stream>>>(mean1, 256, h0, 256, A1p, 16, 4096);
  gemm_kernel<8, true><<<dim3(64, 2), 128, 0, stream>>>(
      A1p, B1p, bl1, h1, 256, 16, 16);

  // ---- output projection ----
  // A2 = h1 : M=4096, K=256 -> KT=8, tiles=256*8=2048
  packA_kernel<<<256, 256, 0, stream>>>(h1, 256, h1, 0, A2p, 8, 2048);
  gemm_kernel<3, false><<<dim3(64, 1), 128, 0, stream>>>(
      A2p, B2p, bo, out, 47, 8, 3);
}
